// BERT_TreeLSTM_BiLSTM_CNN_JointModel_43508018708846
// MI455X (gfx1250) — matrix operand order repaired and checked by reading, not compiled, not run on hardware
//
#include <hip/hip_runtime.h>
#include <hip/hip_bf16.h>

// ---------------- model dims ----------------
#define SEQ   1024
#define DIN   620      // EMB(300)+CHAR_ENC(128)+3*POS_SZ(192)
#define KPAD  640      // 620 padded to multiple of 32 for WMMA K loop
#define BIH   128
#define G4    512      // 4*BIH
#define TLH   256
#define IOU3  768      // 3*TLH
#define LCC   128
#define GCC   256
#define TOKF  640      // 2*BIH + LCC + TLH
#define SENTF 1536     // 2*TOKF + GCC
#define NTAGS 16
#define NREL  16
#define NET   8

typedef __attribute__((ext_vector_type(16))) __bf16   v16bf;
typedef __attribute__((ext_vector_type(8)))  float    v8f;
typedef __attribute__((ext_vector_type(4)))  unsigned v4u;

__device__ __forceinline__ float sigf(float x) { return 1.0f / (1.0f + __expf(-x)); }

__device__ __forceinline__ unsigned short f2bfu(float f) {
  unsigned u = __float_as_uint(f);
  unsigned r = u + 0x7FFFu + ((u >> 16) & 1u);   // round-to-nearest-even
  return (unsigned short)(r >> 16);
}

// ---------------- generic zero ----------------
__global__ void zero_kernel(float* p, long long n) {
  long long i = (long long)blockIdx.x * blockDim.x + threadIdx.x;
  if (i < n) p[i] = 0.0f;
}

// ---------------- embeddings + char-CNN -> x[SEQ][DIN] ----------------
__global__ void build_inputs_kernel(const int* __restrict__ wid, const int* __restrict__ cid,
                                    const int* __restrict__ pid, const int* __restrict__ did,
                                    const int* __restrict__ posid,
                                    const float* __restrict__ wtab, const float* __restrict__ ctab,
                                    const float* __restrict__ ccw, const float* __restrict__ ccb,
                                    const float* __restrict__ postab, const float* __restrict__ ptab,
                                    float* __restrict__ X) {
  __shared__ float cemb[16 * 50];
  int t = blockIdx.x;
  int tid = threadIdx.x;                 // 128 threads
  for (int idx = tid; idx < 16 * 50; idx += 128) {
    int w = idx / 50, i = idx % 50;
    cemb[idx] = ctab[(long long)cid[t * 16 + w] * 50 + i];
  }
  __syncthreads();
  long long wrow = (long long)wid[t] * 300;
  for (int j = tid; j < 300; j += 128) X[(long long)t * DIN + j] = wtab[wrow + j];
  if (tid < 64) {
    X[(long long)t * DIN + 428 + tid] = postab[(long long)pid[t]   * 64 + tid];
    X[(long long)t * DIN + 492 + tid] = ptab  [(long long)posid[t] * 64 + tid];
    X[(long long)t * DIN + 556 + tid] = ptab  [(long long)did[t]   * 64 + tid];
  }
  // char conv (k=3, pad=1) + relu + max over width -> 128 channels
  int o = tid;
  float best = -3.4e38f;
  for (int w = 0; w < 16; ++w) {
    float acc = ccb[o];
    #pragma unroll
    for (int dt = 0; dt < 3; ++dt) {
      int ww = w + dt - 1;
      if (ww < 0 || ww >= 16) continue;
      const float* cw = ccw + (long long)o * 150 + dt;
      const float* ce = cemb + ww * 50;
      for (int i = 0; i < 50; ++i) acc += ce[i] * cw[i * 3];
    }
    best = fmaxf(best, acc);
  }
  X[(long long)t * DIN + 300 + o] = fmaxf(best, 0.0f);
}

// ---- A conversion: f32[Rsrc x Csrc] -> bf16[Rdst x Cdst], src row = r + rowOff ----
// Guard rows (r+rowOff outside [0,Rsrc)) and pad cols are written as zero.
__global__ void convert_A(const float* __restrict__ src, unsigned short* __restrict__ dst,
                          int Rdst, int Csrc, int Cdst, int Rsrc, int rowOff) {
  long long i = (long long)blockIdx.x * blockDim.x + threadIdx.x;
  long long tot = (long long)Rdst * Cdst;
  if (i >= tot) return;
  int r = (int)(i / Cdst), c = (int)(i % Cdst);
  int sr = r + rowOff;
  unsigned short v = 0;
  if (sr >= 0 && sr < Rsrc && c < Csrc) v = f2bfu(src[(long long)sr * Csrc + c]);
  dst[i] = v;
}

// ---- weight conversion with transpose: W[K x N] f32 -> Bt[N x Kp] bf16 (K padded) ----
__global__ void convert_wT(const float* __restrict__ src, unsigned short* __restrict__ dst,
                           int Ksrc, int N, int Kp) {
  long long i = (long long)blockIdx.x * blockDim.x + threadIdx.x;
  long long tot = (long long)N * Kp;
  if (i >= tot) return;
  int n = (int)(i / Kp), k = (int)(i % Kp);
  unsigned short v = 0;
  if (k < Ksrc) v = f2bfu(src[(long long)k * N + n]);
  dst[i] = v;
}

// ---- conv weights (OC, Din, 3) -> 3 transposed matrices [OC x Kp] bf16 ----
__global__ void convert_conv_wT(const float* __restrict__ w, unsigned short* __restrict__ dst,
                                int OC, int Din, int Kp) {
  long long i = (long long)blockIdx.x * blockDim.x + threadIdx.x;
  long long per = (long long)OC * Kp;
  if (i >= 3 * per) return;
  int dt = (int)(i / per);
  long long rem = i - (long long)dt * per;
  int o = (int)(rem / Kp), k = (int)(rem % Kp);
  unsigned short v = 0;
  if (k < Din) v = f2bfu(w[((long long)o * Din + k) * 3 + dt]);
  dst[i] = v;
}

// ---------------- WMMA bf16 GEMM, register-blocked TMx x TNx 16-tiles per wave ------
// A: row-major [M x K] bf16 with one zeroed guard row before row 0 and after row M-1
//    (pointer passed is logical row 0). shift in [-1,1] therefore never leaves storage.
// Bt: [N x K] bf16 (B transposed) so each lane's fragment is two 16-byte runs along K.
// C: row-major f32 [M x N].
template <int TMX, int TNX>
__global__ void wmma_gemm_bf16(const unsigned short* __restrict__ A,
                               const unsigned short* __restrict__ Bt,
                               float* __restrict__ C,
                               int M, int N, int K, int shift,
                               const float* __restrict__ bias, int accumulate) {
  int lane = threadIdx.x & 31;
  int half = lane >> 4;
  int l15  = lane & 15;
  int bn = blockIdx.x * TNX;             // first 16-tile in N
  int bm = blockIdx.y * TMX;             // first 16-tile in M

  // fragment base pointers (guard rows make rA always addressable)
  const unsigned short* aptr[TMX];
  #pragma unroll
  for (int i = 0; i < TMX; ++i) {
    int rA = (bm + i) * 16 + l15 + shift;            // in [-1, M]
    aptr[i] = A + (long long)rA * K + half * 8;      // ISA A layout: K-half select
  }
  const unsigned short* bptr[TNX];
  #pragma unroll
  for (int j = 0; j < TNX; ++j) {
    int nB = (bn + j) * 16 + l15;
    bptr[j] = Bt + (long long)nB * K + half * 8;     // ISA B layout: K-half select
  }

  union F8 { v8f v; float f[8]; };
  F8 c[TMX][TNX];
  #pragma unroll
  for (int i = 0; i < TMX; ++i)
    #pragma unroll
    for (int j = 0; j < TNX; ++j) {
      int cm0 = (bm + i) * 16 + half * 8;
      int cn  = (bn + j) * 16 + l15;
      if (accumulate) {
        #pragma unroll
        for (int r = 0; r < 8; ++r) c[i][j].f[r] = C[(long long)(cm0 + r) * N + cn];
      } else {
        float bv = bias ? bias[cn] : 0.0f;
        #pragma unroll
        for (int r = 0; r < 8; ++r) c[i][j].f[r] = bv;
      }
    }

  union AB { v16bf v; v4u q[2]; };
  for (int k = 0; k < K; k += 32) {
    AB a[TMX], b[TNX];
    #pragma unroll
    for (int i = 0; i < TMX; ++i) {                  // two b128 loads per fragment
      a[i].q[0] = *(const v4u*)(aptr[i] + k);        // K = half*8 .. +7
      a[i].q[1] = *(const v4u*)(aptr[i] + k + 16);   // K = 16+half*8 .. +7
    }
    #pragma unroll
    for (int j = 0; j < TNX; ++j) {
      b[j].q[0] = *(const v4u*)(bptr[j] + k);        // K = half*8 .. +7
      b[j].q[1] = *(const v4u*)(bptr[j] + k + 16);   // K = 16+half*8 .. +7
    }
    #pragma unroll
    for (int i = 0; i < TMX; ++i)
      #pragma unroll
      for (int j = 0; j < TNX; ++j)
        c[i][j].v = __builtin_amdgcn_wmma_f32_16x16x32_bf16(false, a[i].v, false, b[j].v,
                                                            (short)0, c[i][j].v, false, false);
  }

  #pragma unroll
  for (int i = 0; i < TMX; ++i)
    #pragma unroll
    for (int j = 0; j < TNX; ++j) {
      int cm0 = (bm + i) * 16 + half * 8;
      int cn  = (bn + j) * 16 + l15;
      #pragma unroll
      for (int r = 0; r < 8; ++r) C[(long long)(cm0 + r) * N + cn] = c[i][j].f[r];
    }
}

// ---------------- BiLSTM scans (one workgroup per direction) ----------------
__global__ void lstm_kernel(const float* __restrict__ XWf, const float* __restrict__ XWb,
                            const float* __restrict__ Whf, const float* __restrict__ Whb,
                            float* __restrict__ HF, float* __restrict__ HB) {
  __shared__ float h[BIH], cc[BIH], g[G4];
  int dir = blockIdx.x;
  const float* XW = dir ? XWb : XWf;
  const float* Wh = dir ? Whb : Whf;
  float* H = dir ? HB : HF;
  int n = threadIdx.x;                     // 512 threads
  if (n < BIH) { h[n] = 0.0f; cc[n] = 0.0f; }
  __syncthreads();
  for (int t = 0; t < SEQ; ++t) {
    int row = dir ? (SEQ - 1 - t) : t;
    float acc = XW[(long long)row * G4 + n];        // x@Wx + b precomputed via WMMA
    for (int k = 0; k < BIH; ++k) acc += h[k] * Wh[(long long)k * G4 + n];
    g[n] = acc;
    __syncthreads();
    if (n < BIH) {
      float i = g[n], f = g[BIH + n], gg = g[2 * BIH + n], o = g[3 * BIH + n];
      float cn = sigf(f) * cc[n] + sigf(i) * tanhf(gg);
      float hn = sigf(o) * tanhf(cn);
      cc[n] = cn; h[n] = hn;
      H[(long long)row * BIH + n] = hn;
    }
    __syncthreads();
  }
}

// ---------------- Tree-LSTM scan (single workgroup) ----------------
__global__ void tree_lstm_kernel(const float* __restrict__ XIOU, const float* __restrict__ XF,
                                 const float* __restrict__ Uiou, const float* __restrict__ Uf,
                                 const int* __restrict__ order, const int* __restrict__ parent,
                                 float* __restrict__ HSUM, float* __restrict__ FCSUM,
                                 float* __restrict__ TRH) {
  __shared__ float hs[TLH], iou[IOU3], hj[TLH], cj[TLH];
  int n = threadIdx.x;                     // 768 threads
  for (int jj = 0; jj < SEQ; ++jj) {
    int j = order[jj];
    if (n < TLH) hs[n] = HSUM[(long long)j * TLH + n];
    __syncthreads();
    float acc = XIOU[(long long)j * IOU3 + n];      // x@Wiou + b precomputed via WMMA
    for (int k = 0; k < TLH; ++k) acc += hs[k] * Uiou[(long long)k * IOU3 + n];
    iou[n] = acc;
    __syncthreads();
    if (n < TLH) {
      float i = iou[n], o = iou[TLH + n], u = iou[2 * TLH + n];
      float c = sigf(i) * tanhf(u) + FCSUM[(long long)j * TLH + n];
      float h = sigf(o) * tanhf(c);
      cj[n] = c; hj[n] = h;
      TRH[(long long)j * TLH + n] = h;
    }
    __syncthreads();
    if (n < TLH) {
      int p = parent[j];
      float a2 = XF[(long long)p * TLH + n];        // x@Wf + b (row SEQ == zeros)
      for (int k = 0; k < TLH; ++k) a2 += hj[k] * Uf[(long long)k * TLH + n];
      float f = sigf(a2);
      HSUM [(long long)p * TLH + n] += hj[n];
      FCSUM[(long long)p * TLH + n] += f * cj[n];
    }
    __syncthreads();
  }
}

// ---------------- column-wise max over rows (global CNN pooling) ----------------
__global__ void colmax_kernel(const float* __restrict__ X, float* __restrict__ out,
                              int rows, int cols) {
  int c = blockIdx.x * blockDim.x + threadIdx.x;
  if (c >= cols) return;
  float m = -3.4e38f;
  for (int r = 0; r < rows; ++r) m = fmaxf(m, X[(long long)r * cols + c]);
  out[c] = m;
}

// ---------------- assemble sentence_encoding (f32 out + bf16 copy) ----------------
__device__ __forceinline__ float tokfeat(int t, int c,
                                         const float* HF, const float* HB,
                                         const float* LOC, const float* TRH) {
  if (c < 128) return HF[(long long)t * 128 + c];
  if (c < 256) return HB[(long long)t * 128 + (c - 128)];
  if (c < 384) return LOC[(long long)t * 128 + (c - 256)];
  return TRH[(long long)t * 256 + (c - 384)];
}

__global__ void assemble_kernel(const float* __restrict__ HF, const float* __restrict__ HB,
                                const float* __restrict__ LOC, const float* __restrict__ TRH,
                                const float* __restrict__ GMAX, const int* __restrict__ ptr_idx,
                                float* __restrict__ SE, unsigned short* __restrict__ SEbf) {
  int t = blockIdx.x;
  int p = *ptr_idx;
  for (int c = threadIdx.x; c < SENTF; c += blockDim.x) {
    float v;
    if (c < TOKF)            v = tokfeat(t, c, HF, HB, LOC, TRH);
    else if (c < 2 * TOKF)   v = tokfeat(p, c - TOKF, HF, HB, LOC, TRH);
    else                     v = GMAX[c - 2 * TOKF];
    SE  [(long long)t * SENTF + c] = v;
    SEbf[(long long)t * SENTF + c] = f2bfu(v);      // SEbf points at logical row 0
  }
}

// ---------------- elementwise sigmoid ----------------
__global__ void sigmoid_kernel(const float* __restrict__ in, float* __restrict__ out, int n) {
  int i = blockIdx.x * blockDim.x + threadIdx.x;
  if (i < n) out[i] = sigf(in[i]);
}

// ---------------- Viterbi over 16 tags ----------------
__global__ void viterbi_kernel(const float* __restrict__ emit, const float* __restrict__ trans,
                               int* __restrict__ ptrs, float* __restrict__ ent_out) {
  __shared__ float score[NTAGS], ns[NTAGS];
  int j = threadIdx.x;
  if (j < NTAGS) score[j] = emit[j];
  __syncthreads();
  for (int t = 1; t < SEQ; ++t) {
    if (j < NTAGS) {
      float best = -3.4e38f; int bi = 0;
      for (int i = 0; i < NTAGS; ++i) {
        float s = score[i] + trans[i * NTAGS + j];
        if (s > best) { best = s; bi = i; }
      }
      ptrs[t * NTAGS + j] = bi;
      ns[j] = best + emit[t * NTAGS + j];
    }
    __syncthreads();
    if (j < NTAGS) score[j] = ns[j];
    __syncthreads();
  }
  if (j == 0) {
    int last = 0; float b = score[0];
    for (int i = 1; i < NTAGS; ++i) if (score[i] > b) { b = score[i]; last = i; }
    int tag = last;
    ent_out[SEQ - 1] = (float)last;
    for (int t = SEQ - 1; t >= 1; --t) {
      int prev = ptrs[t * NTAGS + tag];
      ent_out[t - 1] = (float)prev;
      tag = prev;
    }
  }
}

// ---------------- entity-type head: softmax((gmax ++ ptr) @ W + b) ----------------
__global__ void entitytype_kernel(const float* __restrict__ GMAX, const float* __restrict__ SE,
                                  const int* __restrict__ ptr_idx,
                                  const float* __restrict__ etw, const float* __restrict__ etb,
                                  float* __restrict__ out) {
  __shared__ float lg[NET];
  int r = threadIdx.x;
  int p = *ptr_idx;
  if (r < NET) {
    float acc = etb[r];
    for (int k = 0; k < GCC; ++k) acc += GMAX[k] * etw[k * NET + r];
    const float* pr = SE + (long long)p * SENTF;   // first TOKF cols of row p == tokens_info[p]
    for (int k = 0; k < TOKF; ++k) acc += pr[k] * etw[(GCC + k) * NET + r];
    lg[r] = acc;
  }
  __syncthreads();
  if (r == 0) {
    float m = lg[0];
    for (int i = 1; i < NET; ++i) m = fmaxf(m, lg[i]);
    float s = 0.0f, e[NET];
    for (int i = 0; i < NET; ++i) { e[i] = __expf(lg[i] - m); s += e[i]; }
    for (int i = 0; i < NET; ++i) out[i] = e[i] / s;
  }
}

// ============================================================================
extern "C" void kernel_launch(void* const* d_in, const int* in_sizes, int n_in,
                              void* d_out, int out_size, void* d_ws, size_t ws_size,
                              hipStream_t stream) {
  const int*   wid    = (const int*)  d_in[0];
  const int*   cid    = (const int*)  d_in[1];
  const int*   pid    = (const int*)  d_in[2];
  const int*   did    = (const int*)  d_in[3];
  const int*   posid  = (const int*)  d_in[4];
  const int*   order  = (const int*)  d_in[5];
  const int*   parent = (const int*)  d_in[6];
  const int*   ptrIdx = (const int*)  d_in[7];
  const float* wtab   = (const float*)d_in[8];
  const float* ctab   = (const float*)d_in[9];
  const float* ccw    = (const float*)d_in[10];
  const float* ccb    = (const float*)d_in[11];
  const float* postab = (const float*)d_in[12];
  const float* ptab   = (const float*)d_in[13];
  const float* Wxf    = (const float*)d_in[14];
  const float* Whf    = (const float*)d_in[15];
  const float* bLf    = (const float*)d_in[16];
  const float* Wxb    = (const float*)d_in[17];
  const float* Whb    = (const float*)d_in[18];
  const float* bLb    = (const float*)d_in[19];
  const float* wcw    = (const float*)d_in[20];
  const float* wcb    = (const float*)d_in[21];
  const float* Wiou   = (const float*)d_in[22];
  const float* Uiou   = (const float*)d_in[23];
  const float* biou   = (const float*)d_in[24];
  const float* Wft    = (const float*)d_in[25];
  const float* Uf     = (const float*)d_in[26];
  const float* bft    = (const float*)d_in[27];
  const float* scw    = (const float*)d_in[28];
  const float* scb    = (const float*)d_in[29];
  const float* etw    = (const float*)d_in[30];
  const float* etb    = (const float*)d_in[31];
  const float* crfw   = (const float*)d_in[32];
  const float* crfb   = (const float*)d_in[33];
  const float* crft   = (const float*)d_in[34];
  const float* relw   = (const float*)d_in[35];
  const float* relb   = (const float*)d_in[36];
  float* out = (float*)d_out;

  // ---- workspace bump allocator ----
  char* base = (char*)d_ws; size_t off = 0;
  auto alloc = [&](size_t bytes) -> void* {
    void* p = base + off; off = (off + bytes + 255) & ~(size_t)255; return p;
  };
  float* X = (float*)alloc((size_t)SEQ * DIN * 4);
  unsigned short* AbfG  = (unsigned short*)alloc((size_t)(SEQ + 2) * KPAD * 2);  // guard rows
  unsigned short* Abf   = AbfG + KPAD;                                           // logical row 0
  unsigned short* bWxf  = (unsigned short*)alloc((size_t)G4   * KPAD * 2);   // [N][K] transposed
  unsigned short* bWxb  = (unsigned short*)alloc((size_t)G4   * KPAD * 2);
  unsigned short* bWiou = (unsigned short*)alloc((size_t)IOU3 * KPAD * 2);
  unsigned short* bWf   = (unsigned short*)alloc((size_t)TLH  * KPAD * 2);
  unsigned short* bWloc = (unsigned short*)alloc((size_t)3 * LCC * KPAD * 2);
  unsigned short* bWsnt = (unsigned short*)alloc((size_t)3 * GCC * KPAD * 2);
  unsigned short* bWcrf = (unsigned short*)alloc((size_t)NTAGS * SENTF * 2);
  unsigned short* bWrel = (unsigned short*)alloc((size_t)NREL  * SENTF * 2);
  float* XWf   = (float*)alloc((size_t)SEQ * G4 * 4);
  float* XWb   = (float*)alloc((size_t)SEQ * G4 * 4);
  float* XIOU  = (float*)alloc((size_t)SEQ * IOU3 * 4);
  float* XF    = (float*)alloc((size_t)(SEQ + 1) * TLH * 4);
  float* HF    = (float*)alloc((size_t)SEQ * BIH * 4);
  float* HB    = (float*)alloc((size_t)SEQ * BIH * 4);
  float* LOC   = (float*)alloc((size_t)SEQ * LCC * 4);
  float* TRH   = (float*)alloc((size_t)SEQ * TLH * 4);
  float* SENTC = (float*)alloc((size_t)SEQ * GCC * 4);
  float* GMAX  = (float*)alloc((size_t)GCC * 4);
  float* HSUM  = (float*)alloc((size_t)(SEQ + 1) * TLH * 4);
  float* FCSUM = (float*)alloc((size_t)(SEQ + 1) * TLH * 4);
  unsigned short* SEbfG = (unsigned short*)alloc((size_t)(SEQ + 2) * SENTF * 2); // guard rows
  unsigned short* SEbf  = SEbfG + SENTF;                                         // logical row 0
  float* EMIT  = (float*)alloc((size_t)SEQ * NTAGS * 4);
  float* RELT  = (float*)alloc((size_t)SEQ * NREL * 4);
  int*   PTRS  = (int*)  alloc((size_t)SEQ * NTAGS * 4);

  const long long OFF_ET  = (long long)SEQ * SENTF;
  const long long OFF_ENT = OFF_ET + NET;
  const long long OFF_REL = OFF_ENT + SEQ;

  auto cdiv = [](long long a, long long b) { return (int)((a + b - 1) / b); };
  // 2x2 16-tiles per wave (32x32 C per wave): for the wide GEMMs
  auto gemm22 = [&](const unsigned short* A, const unsigned short* Bt, float* C,
                    int M, int N, int K, int shift, const float* bias, int acc) {
    dim3 g(N / 32, M / 32);
    wmma_gemm_bf16<2, 2><<<g, 32, 0, stream>>>(A, Bt, C, M, N, K, shift, bias, acc);
  };
  // 2x1 16-tiles per wave: for the N=16 head GEMMs
  auto gemm21 = [&](const unsigned short* A, const unsigned short* Bt, float* C,
                    int M, int N, int K, int shift, const float* bias, int acc) {
    dim3 g(N / 16, M / 32);
    wmma_gemm_bf16<2, 1><<<g, 32, 0, stream>>>(A, Bt, C, M, N, K, shift, bias, acc);
  };

  // 1) inputs + char CNN
  build_inputs_kernel<<<SEQ, 128, 0, stream>>>(wid, cid, pid, did, posid,
                                               wtab, ctab, ccw, ccb, postab, ptab, X);
  // 2) bf16 conversions (A row-major with guard rows; weights transposed to [N][K])
  convert_A<<<cdiv((long long)(SEQ + 2) * KPAD, 256), 256, 0, stream>>>(
      X, AbfG, SEQ + 2, DIN, KPAD, SEQ, -1);
  convert_wT<<<cdiv((long long)G4   * KPAD, 256), 256, 0, stream>>>(Wxf,  bWxf,  DIN, G4,   KPAD);
  convert_wT<<<cdiv((long long)G4   * KPAD, 256), 256, 0, stream>>>(Wxb,  bWxb,  DIN, G4,   KPAD);
  convert_wT<<<cdiv((long long)IOU3 * KPAD, 256), 256, 0, stream>>>(Wiou, bWiou, DIN, IOU3, KPAD);
  convert_wT<<<cdiv((long long)TLH  * KPAD, 256), 256, 0, stream>>>(Wft,  bWf,   DIN, TLH,  KPAD);
  convert_wT<<<cdiv((long long)NTAGS * SENTF, 256), 256, 0, stream>>>(crfw, bWcrf, SENTF, NTAGS, SENTF);
  convert_wT<<<cdiv((long long)NREL  * SENTF, 256), 256, 0, stream>>>(relw, bWrel, SENTF, NREL,  SENTF);
  convert_conv_wT<<<cdiv((long long)3 * LCC * KPAD, 256), 256, 0, stream>>>(wcw, bWloc, LCC, DIN, KPAD);
  convert_conv_wT<<<cdiv((long long)3 * GCC * KPAD, 256), 256, 0, stream>>>(scw, bWsnt, GCC, DIN, KPAD);

  // 3) batched WMMA GEMMs (hoisted out of the scans)
  gemm22(Abf, bWxf,  XWf,  SEQ, G4,   KPAD, 0, bLf,  0);
  gemm22(Abf, bWxb,  XWb,  SEQ, G4,   KPAD, 0, bLb,  0);
  gemm22(Abf, bWiou, XIOU, SEQ, IOU3, KPAD, 0, biou, 0);
  gemm22(Abf, bWf,   XF,   SEQ, TLH,  KPAD, 0, bft,  0);
  zero_kernel<<<1, 256, 0, stream>>>(XF + (long long)SEQ * TLH, TLH);   // xf row S = 0
  for (int dt = 0; dt < 3; ++dt)   // conv(k=3,pad=1) == 3 shifted GEMMs
    gemm22(Abf, bWloc + (size_t)dt * LCC * KPAD, LOC,   SEQ, LCC, KPAD, dt - 1, wcb, dt == 0 ? 0 : 1);
  for (int dt = 0; dt < 3; ++dt)
    gemm22(Abf, bWsnt + (size_t)dt * GCC * KPAD, SENTC, SEQ, GCC, KPAD, dt - 1, scb, dt == 0 ? 0 : 1);

  // 4) sequential recurrences
  lstm_kernel<<<2, G4, 0, stream>>>(XWf, XWb, Whf, Whb, HF, HB);
  zero_kernel<<<cdiv((long long)(SEQ + 1) * TLH, 256), 256, 0, stream>>>(HSUM,  (long long)(SEQ + 1) * TLH);
  zero_kernel<<<cdiv((long long)(SEQ + 1) * TLH, 256), 256, 0, stream>>>(FCSUM, (long long)(SEQ + 1) * TLH);
  tree_lstm_kernel<<<1, IOU3, 0, stream>>>(XIOU, XF, Uiou, Uf, order, parent, HSUM, FCSUM, TRH);

  // 5) pooling + assembly (also zero SEbf guard rows: SENTF ushorts == SENTF/2 floats)
  colmax_kernel<<<1, GCC, 0, stream>>>(SENTC, GMAX, SEQ, GCC);
  zero_kernel<<<cdiv(SENTF / 2, 256), 256, 0, stream>>>((float*)SEbfG, SENTF / 2);
  zero_kernel<<<cdiv(SENTF / 2, 256), 256, 0, stream>>>(
      (float*)(SEbfG + (long long)(SEQ + 1) * SENTF), SENTF / 2);
  assemble_kernel<<<SEQ, 256, 0, stream>>>(HF, HB, LOC, TRH, GMAX, ptrIdx, out, SEbf);

  // 6) heads (WMMA GEMMs over sentence encoding) + small kernels
  gemm21(SEbf, bWcrf, EMIT, SEQ, NTAGS, SENTF, 0, crfb, 0);
  gemm21(SEbf, bWrel, RELT, SEQ, NREL,  SENTF, 0, relb, 0);
  sigmoid_kernel<<<cdiv((long long)SEQ * NREL, 256), 256, 0, stream>>>(RELT, out + OFF_REL, SEQ * NREL);
  viterbi_kernel<<<1, 32, 0, stream>>>(EMIT, crft, PTRS, out + OFF_ENT);
  entitytype_kernel<<<1, 32, 0, stream>>>(GMAX, out, ptrIdx, etw, etb, out + OFF_ET);
}
